// AgentRNN_26628797235781
// MI455X (gfx1250) — matrix operand order repaired and checked by reading, not compiled, not run on hardware
//
#include <hip/hip_runtime.h>
#include <hip/hip_bf16.h>

// ---------------------------------------------------------------------------
// AgentRNN (GRU rollout) for gfx1250 / MI455X.
//  T=512, B=128, OBS=512, H=1024, A=64.
//  All GEMMs use v_wmma_f32_16x16x32_bf16 with f32 accumulation.
//  B tiles are cooperatively staged through LDS with explicit double
//  buffering (ping-pong + barrier); ds reads for all tiles are batched ahead
//  of the WMMA block so dscnt waits are partial. A tiles stay in a per-wave
//  register pipeline. The next k-step's global loads overlap the WMMAs.
// ---------------------------------------------------------------------------

#define TT 512
#define BB 128
#define OBS 512
#define HH 1024
#define AA 64
#define MM (TT * BB)        // 65536 rows for the batched GEMMs

typedef __attribute__((ext_vector_type(16))) __bf16 v16bf;
typedef __attribute__((ext_vector_type(8)))  float  v8f;

// ---- helpers ---------------------------------------------------------------

__device__ __forceinline__ unsigned short f2bfbits_rne(float f) {
    unsigned u = __builtin_bit_cast(unsigned, f);
    return (unsigned short)((u + 0x7FFFu + ((u >> 16) & 1u)) >> 16);
}
__device__ __forceinline__ unsigned short f2bfbits_trunc(float f) {
    return (unsigned short)(__builtin_bit_cast(unsigned, f) >> 16);
}
__device__ __forceinline__ unsigned pack_bf16x2(float f0, float f1) {
    return __builtin_amdgcn_perm(__builtin_bit_cast(unsigned, f1),
                                 __builtin_bit_cast(unsigned, f0), 0x07060302u);
}

__device__ __forceinline__ v8f wmma_bf16(v16bf a, v16bf b, v8f c) {
    return __builtin_amdgcn_wmma_f32_16x16x32_bf16(
        false, a, false, b, (short)0, c, false, false);
}

// A-tile (16x32 bf16) from row-major bf16 matrix, with dword AND-mask
// (mask = 0 zeroes the row -> GRU done-reset; ~0 = passthrough).
__device__ __forceinline__ v16bf load_a_bf16(const unsigned short* __restrict__ A,
                                             int row, int ld, int k0, int lane,
                                             unsigned msk) {
    int off = (lane >= 16) ? 16 : 0;
    const unsigned* p = reinterpret_cast<const unsigned*>(A + (size_t)row * ld + k0 + off);
    uint4 lo = *reinterpret_cast<const uint4*>(p);
    uint4 hi = *reinterpret_cast<const uint4*>(p + 4);
    union { uint4 q[2]; v16bf v; } u;
    u.q[0] = make_uint4(lo.x & msk, lo.y & msk, lo.z & msk, lo.w & msk);
    u.q[1] = make_uint4(hi.x & msk, hi.y & msk, hi.z & msk, hi.w & msk);
    return u.v;
}

// B-tile (32x16 bf16) read from LDS: one contiguous 32B ds read per lane.
__device__ __forceinline__ v16bf lds_b(const unsigned short* buf, int tile, int lane) {
    return *reinterpret_cast<const v16bf*>(buf + tile * 512 + lane * 16);
}

__device__ __forceinline__ float sigmoidf_(float x) {
    return 1.0f / (1.0f + __expf(-x));
}

// ---- one-time f32 -> bf16 convert ------------------------------------------
__global__ __launch_bounds__(256) void cvt_f32_bf16_kernel(const float* __restrict__ src,
                                                           unsigned* __restrict__ dst,
                                                           int npair) {
    int i = blockIdx.x * blockDim.x + threadIdx.x;
    if (i >= npair) return;
    float2 f = reinterpret_cast<const float2*>(src)[i];
    dst[i] = pack_bf16x2(f.x, f.y);
}

// ---- weight packing: f32 row-major [K,N] -> bf16 WMMA B-layout -------------
// Per (kb,nb) tile: 32 lanes x 16 halfs. Lane L holds column n = nb*16 + L%16,
// halfs j map to K = kb*32 + j + 16*(L/16).
__global__ __launch_bounds__(256) void pack_w_kernel(const float* __restrict__ W,
                                                     unsigned short* __restrict__ out,
                                                     int K, int N) {
    int idx  = blockIdx.x * blockDim.x + threadIdx.x;
    int lane = idx & 31;
    int tile = idx >> 5;
    int NT = N >> 4, KT = K >> 5;
    if (tile >= KT * NT) return;
    int kb = tile / NT, nb = tile % NT;
    int n = nb * 16 + (lane & 15);
    int kbase = kb * 32 + ((lane >= 16) ? 16 : 0);
    unsigned short* o = out + (size_t)tile * 512 + lane * 16;
#pragma unroll
    for (int j = 0; j < 16; ++j)
        o[j] = f2bfbits_rne(W[(size_t)(kbase + j) * N + n]);
}

// ---- generic LDS-staged GEMM body ------------------------------------------
// Block = 256 threads = 8 waves. All waves share one group of NBLK=4 column
// tiles (4KB of B per k-step, staged by 256x b128) and take 8 distinct M
// tiles. STORE_MODE: 0 = f32 + bias ; 1 = bf16 + bias + relu.
template <int K, int N, int STORE_MODE>
__device__ __forceinline__ void gemm_body(const unsigned short* __restrict__ Abf,
                                          const unsigned short* __restrict__ Wp,
                                          const float* __restrict__ bias,
                                          void* __restrict__ outp,
                                          unsigned short* ldsB /* [2][4*512] */) {
    constexpr int NBLK = 4;
    constexpr int NT = N / 16, NG = NT / NBLK, KB = K / 32;
    int tid  = threadIdx.x;
    int wv   = tid >> 5;
    int lane = tid & 31;
    int ng   = blockIdx.x % NG;
    int mtg  = blockIdx.x / NG;
    int nt0  = ng * NBLK;
    int mt   = mtg * 8 + wv;
    int rowA = mt * 16 + (lane & 15);

    v8f c[NBLK];
#pragma unroll
    for (int j = 0; j < NBLK; ++j) c[j] = (v8f){};

    // prologue: stage kb=0 B (one b128 per thread), load A(0)
    uint4 st = *reinterpret_cast<const uint4*>(Wp + (size_t)nt0 * 512 + tid * 8);
    v16bf a  = load_a_bf16(Abf, rowA, K, 0, lane, 0xFFFFFFFFu);
    *reinterpret_cast<uint4*>(ldsB + tid * 8) = st;
    __syncthreads();

    for (int kb = 0; kb < KB - 1; ++kb) {
        // issue next k-step's global loads first (latency overlapped with WMMA)
        uint4 stn = *reinterpret_cast<const uint4*>(
            Wp + (size_t)((kb + 1) * NT + nt0) * 512 + tid * 8);
        v16bf an = load_a_bf16(Abf, rowA, K, (kb + 1) * 32, lane, 0xFFFFFFFFu);
        // batch all ds reads of the current buffer, then the WMMA block
        const unsigned short* buf = ldsB + (kb & 1) * (NBLK * 512);
        v16bf b[NBLK];
#pragma unroll
        for (int j = 0; j < NBLK; ++j) b[j] = lds_b(buf, j, lane);
#pragma unroll
        for (int j = 0; j < NBLK; ++j) c[j] = wmma_bf16(a, b[j], c[j]);
        // stage next buffer, rotate
        *reinterpret_cast<uint4*>(ldsB + ((kb + 1) & 1) * (NBLK * 512) + tid * 8) = stn;
        a = an;
        __syncthreads();
    }
    {
        const unsigned short* buf = ldsB + ((KB - 1) & 1) * (NBLK * 512);
        v16bf b[NBLK];
#pragma unroll
        for (int j = 0; j < NBLK; ++j) b[j] = lds_b(buf, j, lane);
#pragma unroll
        for (int j = 0; j < NBLK; ++j) c[j] = wmma_bf16(a, b[j], c[j]);
    }

    // epilogue
    int rbase = mt * 16 + ((lane >= 16) ? 8 : 0);
#pragma unroll
    for (int j = 0; j < NBLK; ++j) {
        int col = (nt0 + j) * 16 + (lane & 15);
        float bv = bias[col];
#pragma unroll
        for (int r = 0; r < 8; ++r) {
            float v = c[j][r] + bv;
            if (STORE_MODE == 1) {
                v = v > 0.0f ? v : 0.0f;
                ((unsigned short*)outp)[(size_t)(rbase + r) * N + col] = f2bfbits_trunc(v);
            } else {
                ((float*)outp)[(size_t)(rbase + r) * N + col] = v;
            }
        }
    }
}

// ---- K1: emb = relu(obs_bf16 @ W_emb + b_emb), stored bf16 -----------------
__global__ __launch_bounds__(256) void emb_kernel(const unsigned short* __restrict__ obsb,
                                                  const unsigned short* __restrict__ Wp,
                                                  const float* __restrict__ bias,
                                                  unsigned short* __restrict__ emb) {
    __shared__ unsigned short ldsB[2 * 4 * 512];
    gemm_body<OBS, HH, 1>(obsb, Wp, bias, emb, ldsB);
}

// ---- K2: gi = emb @ Wi + bi (f32) ------------------------------------------
__global__ __launch_bounds__(256) void gi_kernel(const unsigned short* __restrict__ emb,
                                                 const unsigned short* __restrict__ Wp,
                                                 const float* __restrict__ bias,
                                                 float* __restrict__ gi) {
    __shared__ unsigned short ldsB[2 * 4 * 512];
    gemm_body<HH, 3 * HH, 0>(emb, Wp, bias, gi, ldsB);
}

// ---- K4: q = ys @ W_out + b_out (f32, into d_out) --------------------------
__global__ __launch_bounds__(256) void qout_kernel(const unsigned short* __restrict__ ys,
                                                   const unsigned short* __restrict__ Wp,
                                                   const float* __restrict__ bias,
                                                   float* __restrict__ q) {
    __shared__ unsigned short ldsB[2 * 4 * 512];
    gemm_body<HH, AA, 0>(ys, Wp, bias, q, ldsB);
}

// ---- K3: one GRU step ------------------------------------------------------
// Block = 128 threads = 4 waves sharing one H-column tile nt; waves take 4
// M tiles (blockIdx parity picks which half of B=128 rows). The 3 gate
// B-tiles (r/z/n, 3KB) are cooperatively staged through double-buffered LDS.
// A (h_prev) is fed bf16 from the previous step's ys with an AND-mask reset.
__global__ __launch_bounds__(128) void gru_step_kernel(const float* __restrict__ h_in,
                                                       const unsigned short* __restrict__ h_in_bf,
                                                       float* __restrict__ h_out,
                                                       const unsigned short* __restrict__ Whp,
                                                       const float* __restrict__ gi_t,
                                                       const unsigned char* __restrict__ done_t,
                                                       const float* __restrict__ b_hn,
                                                       unsigned short* __restrict__ ys_t) {
    constexpr int K = HH, KB = K / 32;
    constexpr int NT = (3 * HH) / 16;   // 192 column tiles in Wh
    constexpr int HT = HH / 16;         // 64 column tiles per gate
    __shared__ unsigned short ldsB[2 * 3 * 512];

    int tid  = threadIdx.x;
    int wv   = tid >> 5;                          // 0..3
    int lane = tid & 31;
    int nt   = blockIdx.x >> 1;                   // 0..63
    int mt   = ((blockIdx.x & 1) << 2) + wv;      // 0..7
    int rowA = mt * 16 + (lane & 15);
    unsigned msk = done_t[rowA] ? 0u : 0xFFFFFFFFu;

    // cooperative B chunks: 3 tiles * 64 b128-chunks = 192; thread tid takes
    // chunk tid, plus chunk 128+tid when tid < 64.
    int j0 = tid, j1 = tid + 128;
    auto gchunk = [&](int kb, int j) {
        int g = j >> 6, q = j & 63;
        return *reinterpret_cast<const uint4*>(
            Whp + (size_t)(kb * NT + nt + g * HT) * 512 + q * 8);
    };

    v8f cr = {}, cz = {}, cn = {};

    // prologue: stage kb=0, load A(0)
    uint4 t0 = gchunk(0, j0);
    uint4 t1 = (tid < 64) ? gchunk(0, j1) : make_uint4(0, 0, 0, 0);
    v16bf a = load_a_bf16(h_in_bf, rowA, HH, 0, lane, msk);
    *reinterpret_cast<uint4*>(ldsB + j0 * 8) = t0;
    if (tid < 64) *reinterpret_cast<uint4*>(ldsB + j1 * 8) = t1;
    __syncthreads();

    for (int kb = 0; kb < KB - 1; ++kb) {
        uint4 n0 = gchunk(kb + 1, j0);
        uint4 n1 = (tid < 64) ? gchunk(kb + 1, j1) : make_uint4(0, 0, 0, 0);
        v16bf an = load_a_bf16(h_in_bf, rowA, HH, (kb + 1) * 32, lane, msk);

        const unsigned short* buf = ldsB + (kb & 1) * (3 * 512);
        v16bf br = lds_b(buf, 0, lane);
        v16bf bz = lds_b(buf, 1, lane);
        v16bf bn = lds_b(buf, 2, lane);
        cr = wmma_bf16(a, br, cr);
        cz = wmma_bf16(a, bz, cz);
        cn = wmma_bf16(a, bn, cn);

        unsigned short* nbuf = ldsB + ((kb + 1) & 1) * (3 * 512);
        *reinterpret_cast<uint4*>(nbuf + j0 * 8) = n0;
        if (tid < 64) *reinterpret_cast<uint4*>(nbuf + j1 * 8) = n1;
        a = an;
        __syncthreads();
    }
    {
        const unsigned short* buf = ldsB + ((KB - 1) & 1) * (3 * 512);
        v16bf br = lds_b(buf, 0, lane);
        v16bf bz = lds_b(buf, 1, lane);
        v16bf bn = lds_b(buf, 2, lane);
        cr = wmma_bf16(a, br, cr);
        cz = wmma_bf16(a, bz, cz);
        cn = wmma_bf16(a, bn, cn);
    }

    int col   = nt * 16 + (lane & 15);
    float bhn = b_hn[col];
    int rbase = mt * 16 + ((lane >= 16) ? 8 : 0);
#pragma unroll
    for (int r = 0; r < 8; ++r) {
        int row = rbase + r;
        float md = done_t[row] ? 0.0f : 1.0f;
        size_t gbase = (size_t)row * (3 * HH) + col;
        float gir = gi_t[gbase];
        float giz = gi_t[gbase + HH];
        float gin = gi_t[gbase + 2 * HH];
        float rg = sigmoidf_(gir + cr[r]);
        float zg = sigmoidf_(giz + cz[r]);
        float ng = tanhf(gin + rg * (cn[r] + bhn));
        float hm = h_in[(size_t)row * HH + col] * md;   // f32 carry path
        float hnew = (1.0f - zg) * ng + zg * hm;
        h_out[(size_t)row * HH + col] = hnew;
        ys_t[(size_t)row * HH + col] = f2bfbits_trunc(hnew);
    }
}

__global__ __launch_bounds__(256) void copy_f32_kernel(const float* __restrict__ src,
                                                       float* __restrict__ dst, int n) {
    int i = blockIdx.x * blockDim.x + threadIdx.x;
    if (i < n) dst[i] = src[i];
}

// ---------------------------------------------------------------------------
extern "C" void kernel_launch(void* const* d_in, const int* in_sizes, int n_in,
                              void* d_out, int out_size, void* d_ws, size_t ws_size,
                              hipStream_t stream) {
    (void)in_sizes; (void)n_in; (void)out_size; (void)ws_size;

    const float*         hidden = (const float*)d_in[0];           // [B,H]
    const float*         obs    = (const float*)d_in[1];           // [T,B,OBS]
    const unsigned char* dones  = (const unsigned char*)d_in[2];   // [T,B] bool
    const float*         W_emb  = (const float*)d_in[3];           // [OBS,H]
    const float*         b_emb  = (const float*)d_in[4];           // [H]
    const float*         Wi     = (const float*)d_in[5];           // [H,3H]
    const float*         bi     = (const float*)d_in[6];           // [3H]
    const float*         Wh     = (const float*)d_in[7];           // [H,3H]
    const float*         b_hn   = (const float*)d_in[8];           // [H]
    const float*         W_out  = (const float*)d_in[9];           // [H,A]
    const float*         b_out  = (const float*)d_in[10];          // [A]

    // workspace layout
    char* ws = (char*)d_ws;
    size_t off = 0;
    unsigned short* obsb = (unsigned short*)(ws + off); off += (size_t)MM * OBS * 2;      // bf16 obs
    unsigned short* emb  = (unsigned short*)(ws + off); off += (size_t)MM * HH * 2;       // bf16
    float*          gi   = (float*)(ws + off);          off += (size_t)MM * 3 * HH * 4;   // f32
    unsigned short* ys   = (unsigned short*)(ws + off); off += (size_t)MM * HH * 2;       // bf16
    float*          hA   = (float*)(ws + off);          off += (size_t)BB * HH * 4;
    float*          hB   = (float*)(ws + off);          off += (size_t)BB * HH * 4;
    unsigned short* h0b  = (unsigned short*)(ws + off); off += (size_t)BB * HH * 2;       // bf16 init h
    unsigned short* WembP = (unsigned short*)(ws + off); off += (size_t)OBS * HH * 2;
    unsigned short* WiP   = (unsigned short*)(ws + off); off += (size_t)HH * 3 * HH * 2;
    unsigned short* WhP   = (unsigned short*)(ws + off); off += (size_t)HH * 3 * HH * 2;
    unsigned short* WoutP = (unsigned short*)(ws + off); off += (size_t)HH * AA * 2;

    // 0) one-time bf16 conversions
    {
        int np = MM * OBS / 2;
        cvt_f32_bf16_kernel<<<(np + 255) / 256, 256, 0, stream>>>(obs, (unsigned*)obsb, np);
        np = BB * HH / 2;
        cvt_f32_bf16_kernel<<<(np + 255) / 256, 256, 0, stream>>>(hidden, (unsigned*)h0b, np);
    }

    // 1) pack weights into WMMA B layout (bf16)
    {
        int th;
        th = (OBS / 32) * (HH / 16) * 32;
        pack_w_kernel<<<(th + 255) / 256, 256, 0, stream>>>(W_emb, WembP, OBS, HH);
        th = (HH / 32) * (3 * HH / 16) * 32;
        pack_w_kernel<<<(th + 255) / 256, 256, 0, stream>>>(Wi, WiP, HH, 3 * HH);
        pack_w_kernel<<<(th + 255) / 256, 256, 0, stream>>>(Wh, WhP, HH, 3 * HH);
        th = (HH / 32) * (AA / 16) * 32;
        pack_w_kernel<<<(th + 255) / 256, 256, 0, stream>>>(W_out, WoutP, HH, AA);
    }

    // 2) emb = relu(obs @ W_emb + b_emb)   [65536 x 1024]
    {
        int blocks = (MM / 16 / 8) * (HH / 16 / 4);       // 8192
        emb_kernel<<<blocks, 256, 0, stream>>>(obsb, WembP, b_emb, emb);
    }

    // 3) gi = emb @ Wi + bi                [65536 x 3072]
    {
        int blocks = (MM / 16 / 8) * (3 * HH / 16 / 4);   // 24576
        gi_kernel<<<blocks, 256, 0, stream>>>(emb, WiP, bi, gi);
    }

    // 4) sequential GRU scan: 512 stream-ordered step launches (graph-captured)
    const float* hin = hidden;
    const unsigned short* hin_bf = h0b;
    float* hout = hA;
    for (int t = 0; t < TT; ++t) {
        hout = (t & 1) ? hB : hA;
        unsigned short* ys_t = ys + (size_t)t * BB * HH;
        gru_step_kernel<<<128, 128, 0, stream>>>(
            hin, hin_bf, hout, WhP,
            gi + (size_t)t * BB * 3 * HH,
            dones + (size_t)t * BB,
            b_hn, ys_t);
        hin = hout;
        hin_bf = ys_t;      // bf16 h' doubles as next step's A matrix
    }

    // 5) outputs: h_final then q_vals, concatenated flat in d_out
    float* out_h = (float*)d_out;
    float* out_q = out_h + (size_t)BB * HH;
    copy_f32_kernel<<<(BB * HH + 255) / 256, 256, 0, stream>>>(hout, out_h, BB * HH);
    {
        int blocks = (MM / 16 / 8) * (AA / 16 / 4);       // 512
        qout_kernel<<<blocks, 256, 0, stream>>>(ys, WoutP, b_out, out_q);
    }
}
